// UniPhyBlock_64630667870597
// MI455X (gfx1250) — matrix operand order repaired and checked by reading, not compiled, not run on hardware
//
#include <hip/hip_runtime.h>
#include <hip/hip_bf16.h>
#include <math.h>

// ---------------------------------------------------------------------------
// UniPhyBlock for MI455X (gfx1250): wave32 + WMMA f32_16x16x32_f16 everywhere.
// ---------------------------------------------------------------------------

typedef _Float16 half_t;
typedef __attribute__((ext_vector_type(16))) _Float16 v16h;
typedef __attribute__((ext_vector_type(8)))  _Float16 v8h;
typedef __attribute__((ext_vector_type(8)))  float    v8f;

#define DEVFN static __device__ __forceinline__

constexpr int B_  = 4;
constexpr int T_  = 16;
constexpr int D_  = 64;
constexpr int H_  = 64;
constexpr int W_  = 64;
constexpr int D2_ = 128;
constexpr int HW_ = H_ * W_;            // 4096
constexpr int BT_ = B_ * T_;            // 64
constexpr long N5_   = (long)BT_ * D_ * HW_;  // 16,777,216 per r/i tensor
constexpr int  MROWS = B_ * HW_ * T_;         // 262,144 temporal rows
constexpr int  NPIX  = BT_ * HW_;             // 262,144 (b,t,hw) positions
constexpr int  F1_   = 512;                   // D2*EXPAND
constexpr int  KCONV = 9 * D2_;               // 1152
constexpr int  CSTR  = 136;                   // padded channel stride (halfs)

DEVFN v8f wmma_f16(v16h a, v16h b, v8f c) {
  return __builtin_amdgcn_wmma_f32_16x16x32_f16(false, a, false, b,
                                                (short)0, c, false, false);
}

// A fragment (16x32 f16, M=rows): lane m = lane%16, hi = lane/16.
// a[0..7]  = Arow[kt*32 + hi*16 + 0..7]
// a[8..15] = Arow[kt*32 + hi*16 + 8 + 0..7]
DEVFN v16h load_a_frag(const half_t* __restrict__ arow, int hi) {
  v8h lo = *(const v8h*)(arow + hi * 16);
  v8h hh = *(const v8h*)(arow + hi * 16 + 8);
  v16h r;
#pragma unroll
  for (int i = 0; i < 8; ++i) { r[i] = lo[i]; r[8 + i] = hh[i]; }
  return r;
}

DEVFN v16h combine16(v8h lo, v8h hh) {
  v16h r;
#pragma unroll
  for (int i = 0; i < 8; ++i) { r[i] = lo[i]; r[8 + i] = hh[i]; }
  return r;
}

// ---------------------------------------------------------------------------
// Small device-side packing kernels (weights -> WMMA fragment layouts)
// ---------------------------------------------------------------------------

// B fragment layout (32x16 f16, contiguous k order, as in load_a_frag):
// lane: n = lane%16, hi = lane/16 ; element e in [0,16): k = hi*16 + e
__global__ void pack_bfrag(const float* __restrict__ src, half_t* __restrict__ dst,
                           int K, int N) {
  long total = (long)K * N;
  int NT = N >> 4;
  for (long o = (long)blockIdx.x * blockDim.x + threadIdx.x; o < total;
       o += (long)gridDim.x * blockDim.x) {
    int e    = (int)(o & 15);
    int lane = (int)((o >> 4) & 31);
    long tile = o >> 9;
    int nt = (int)(tile % NT);
    int kt = (int)(tile / NT);
    int hi = lane >> 4;
    int k  = kt * 32 + hi * 16 + e;
    int n  = nt * 16 + (lane & 15);
    dst[o] = (half_t)src[(long)k * N + n];
  }
}

// conv weights -> A layout: Wc[c_out][tap*128 + c_in] f16
__global__ void pack_convw(const float* __restrict__ cw, half_t* __restrict__ wc) {
  long total = (long)D2_ * KCONV;
  for (long o = (long)blockIdx.x * blockDim.x + threadIdx.x; o < total;
       o += (long)gridDim.x * blockDim.x) {
    int k   = (int)(o % KCONV);
    int out = (int)(o / KCONV);
    int tap = k >> 7, ci = k & 127;
    int ky = tap / 3, kx = tap % 3;
    wc[o] = (half_t)cw[(((long)out * D2_ + ci) * 3 + ky) * 3 + kx];
  }
}

// W_enc (128x128) = [[Er, Ei], [-Ei, Er]],  W_dec (128x64) = [[Dr], [-Di]]
__global__ void build_wmats(const float* __restrict__ er, const float* __restrict__ ei,
                            const float* __restrict__ dr, const float* __restrict__ di,
                            float* __restrict__ wenc, float* __restrict__ wdec) {
  int idx = blockIdx.x * blockDim.x + threadIdx.x;
  if (idx < D2_ * D2_) {
    int k = idx >> 7, n = idx & 127;
    int kd = k & 63, nd = n & 63;
    float v;
    if (k < 64) v = (n < 64) ? er[kd * 64 + nd] : ei[kd * 64 + nd];
    else        v = (n < 64) ? -ei[kd * 64 + nd] : er[kd * 64 + nd];
    wenc[idx] = v;
  }
  if (idx < D2_ * D_) {
    int k = idx >> 6, n = idx & 63;
    int kd = k & 63;
    wdec[idx] = (k < 64) ? dr[kd * 64 + n] : -di[kd * 64 + n];
  }
}

// ---------------------------------------------------------------------------
// Complex-LN / transpose kernel (shared by stage-1 LN, stage-2 LN, MLP pack)
// out_mode: 0 = f32 same layout (xs), 1 = f16 rows row=(b*HW+hw)*T+t (xt),
//           2 = f16 rows row=bt*HW+hw (xp, no LN)
// ---------------------------------------------------------------------------
__global__ __launch_bounds__(256) void cnorm_kernel(
    const float* __restrict__ in_r, const float* __restrict__ in_i,
    const float* __restrict__ g, const float* __restrict__ bb,
    int do_ln, int out_mode,
    float* __restrict__ out_r, float* __restrict__ out_i,
    half_t* __restrict__ out_rows) {
  __shared__ float lr[64][65];
  __shared__ float li[64][65];
  __shared__ float s1[4][64], s2[4][64];
  __shared__ float mcol[64], iscol[64];

  int btI = blockIdx.x;            // bt (or b*T+t)
  int hw0 = blockIdx.y * 64;
  long base = (long)btI * D_ * HW_ + hw0;

  for (int idx = threadIdx.x; idx < 64 * 64; idx += 256) {
    int d = idx >> 6, c = idx & 63;
    lr[d][c] = in_r[base + (long)d * HW_ + c];
    li[d][c] = in_i[base + (long)d * HW_ + c];
  }
  __syncthreads();

  int col = threadIdx.x & 63, part = threadIdx.x >> 6;
  if (do_ln) {
    float a1 = 0.f, a2 = 0.f;
#pragma unroll
    for (int k = 0; k < 16; ++k) {
      int d = part * 16 + k;
      float r = lr[d][col], im = li[d][col];
      a1 += r + im;
      a2 += r * r + im * im;
    }
    s1[part][col] = a1; s2[part][col] = a2;
    __syncthreads();
    if (threadIdx.x < 64) {
      float t1 = 0.f, t2 = 0.f;
#pragma unroll
      for (int p = 0; p < 4; ++p) { t1 += s1[p][threadIdx.x]; t2 += s2[p][threadIdx.x]; }
      float mean = t1 * (1.f / 128.f);
      float var  = t2 * (1.f / 128.f) - mean * mean;
      mcol[threadIdx.x] = mean;
      iscol[threadIdx.x] = rsqrtf(var + 1e-5f);
    }
    __syncthreads();
  }

  if (out_mode == 0) {
    for (int idx = threadIdx.x; idx < 64 * 64; idx += 256) {
      int d = idx >> 6, c = idx & 63;
      float m = mcol[c], is = iscol[c];
      out_r[base + (long)d * HW_ + c] = (lr[d][c] - m) * is * g[d] + bb[d];
      out_i[base + (long)d * HW_ + c] = (li[d][c] - m) * is * g[64 + d] + bb[64 + d];
    }
  } else {
    int hw = hw0 + col;
    long row;
    if (out_mode == 1) {
      int b = btI >> 4, t = btI & 15;
      row = ((long)(b * HW_ + hw)) * T_ + t;
    } else {
      row = (long)btI * HW_ + hw;
    }
    half_t* dst = out_rows + row * D2_;
    float m  = do_ln ? mcol[col]  : 0.f;
    float is = do_ln ? iscol[col] : 1.f;
#pragma unroll
    for (int k = 0; k < 16; ++k) {
      int d = part * 16 + k;
      float gr = do_ln ? g[d] : 1.f,        br = do_ln ? bb[d] : 0.f;
      float gi = do_ln ? g[64 + d] : 1.f,   bi = do_ln ? bb[64 + d] : 0.f;
      dst[d]      = (half_t)((lr[d][col] - m) * is * gr + br);
      dst[64 + d] = (half_t)((li[d][col] - m) * is * gi + bi);
    }
  }
}

// ---------------------------------------------------------------------------
// Conv 3x3 (128->128 ch) as 9-tap WMMA GEMM. One block per (bt, h) image row:
// M = 128 c_out (wave m-tile), N = 64 w, K = 9*128.
// LDS halo tile: ls[rr][wp][c] with channel-contiguous innermost dim so each
// B fragment is two aligned ds_load_b128 (no divergence, no per-lane branch).
// wp = w+1 (columns 0 and 65 are zero pads for the 'SAME' boundary).
// ---------------------------------------------------------------------------
__global__ __launch_bounds__(256) void conv3x3_wmma(
    const float* __restrict__ xs_r, const float* __restrict__ xs_i,
    const half_t* __restrict__ Wc, const float* __restrict__ conv_b,
    float* __restrict__ cliff_r, float* __restrict__ cliff_i) {
  __shared__ __align__(16) half_t ls[3][W_ + 2][CSTR];   // ~52.6 KB

  int bt = blockIdx.x >> 6;
  int h  = blockIdx.x & 63;

  // stage halo tile (w fastest for coalesced global reads)
  for (int idx = threadIdx.x; idx < 3 * (W_ + 2) * 128; idx += 256) {
    int wp  = idx % (W_ + 2);
    int tmp = idx / (W_ + 2);
    int rr  = tmp % 3;
    int c   = tmp / 3;
    int hh  = h + rr - 1;
    int w   = wp - 1;
    float v = 0.f;
    if (hh >= 0 && hh < H_ && w >= 0 && w < W_) {
      const float* src = (c < 64) ? xs_r : xs_i;
      v = src[((long)(bt * D_ + (c & 63))) * HW_ + hh * W_ + w];
    }
    ls[rr][wp][c] = (half_t)v;
  }
  __syncthreads();

  int lane = threadIdx.x & 31, wid = threadIdx.x >> 5;
  int hi = lane >> 4;
  int m0 = wid * 16;
  v8f acc[4];
#pragma unroll
  for (int i = 0; i < 4; ++i) acc[i] = v8f{};

  const half_t* arow = Wc + (long)(m0 + (lane & 15)) * KCONV;
  for (int tap = 0; tap < 9; ++tap) {
    int rr = (tap >= 6) ? 2 : (tap >= 3) ? 1 : 0;
    int dx = tap - rr * 3;                  // 0..2 ; padded col = wbase + dx
#pragma unroll
    for (int k4 = 0; k4 < 4; ++k4) {
      int kt = tap * 4 + k4;
      v16h a = load_a_frag(arow + kt * 32, hi);
      int cib = k4 * 32 + hi * 16;
#pragma unroll
      for (int nt = 0; nt < 4; ++nt) {
        int wp = nt * 16 + (lane & 15) + dx;
        const half_t* bp = &ls[rr][wp][cib];
        v16h b = combine16(*(const v8h*)bp, *(const v8h*)(bp + 8));
        acc[nt] = wmma_f16(a, b, acc[nt]);
      }
    }
  }

#pragma unroll
  for (int nt = 0; nt < 4; ++nt) {
    int w = nt * 16 + (lane & 15);
#pragma unroll
    for (int v = 0; v < 8; ++v) {
      int co = m0 + v + 8 * hi;
      float val = acc[nt][v] + conv_b[co];
      long idx = ((long)(bt * D_ + (co & 63))) * HW_ + h * W_ + w;
      if (co < 64) cliff_r[idx] = val; else cliff_i[idx] = val;
    }
  }
}

// ---------------------------------------------------------------------------
// 64x64 complex FFT2 * spec_w, inverse FFT2, gate/residual combine (in LDS).
// One block per (bt, d) image.
// ---------------------------------------------------------------------------
DEVFN int rev6(int x) {
  int r = 0;
#pragma unroll
  for (int i = 0; i < 6; ++i) r = (r << 1) | ((x >> i) & 1);
  return r;
}

DEVFN void fft_dim(float* ar, float* ai, int tid, float sgn, int colmode) {
  for (int idx = tid; idx < 4096; idx += 256) {
    int line = colmode ? (idx & 63) : (idx >> 6);
    int p    = colmode ? (idx >> 6) : (idx & 63);
    int rp = rev6(p);
    if (p < rp) {
      int i0 = colmode ? (p  * 64 + line) : (line * 64 + p);
      int i1 = colmode ? (rp * 64 + line) : (line * 64 + rp);
      float tr = ar[i0]; ar[i0] = ar[i1]; ar[i1] = tr;
      float ti = ai[i0]; ai[i0] = ai[i1]; ai[i1] = ti;
    }
  }
  __syncthreads();
  for (int st = 0; st < 6; ++st) {
    int half = 1 << st;
    for (int k = tid; k < 2048; k += 256) {
      int line = colmode ? (k & 63) : (k >> 5);
      int q    = colmode ? (k >> 6) : (k & 31);
      int j = q & (half - 1);
      int base2 = (q >> st) << (st + 1);
      int p0 = base2 + j, p1 = p0 + half;
      int i0 = colmode ? (p0 * 64 + line) : (line * 64 + p0);
      int i1 = colmode ? (p1 * 64 + line) : (line * 64 + p1);
      float ang = sgn * 3.14159265358979323846f * (float)j / (float)half;
      float sn, cs;
      __sincosf(ang, &sn, &cs);
      float xr = ar[i1], xi = ai[i1];
      float tr = cs * xr - sn * xi;
      float ti = cs * xi + sn * xr;
      ar[i1] = ar[i0] - tr; ai[i1] = ai[i0] - ti;
      ar[i0] += tr;         ai[i0] += ti;
    }
    __syncthreads();
  }
}

__global__ __launch_bounds__(256) void fft_spec_combine(
    const float* __restrict__ xs_r, const float* __restrict__ xs_i,
    const float* __restrict__ spec_wr, const float* __restrict__ spec_wi,
    const float* cliff_r_in, const float* cliff_i_in,
    const float* __restrict__ x0_r, const float* __restrict__ x0_i,
    const float* __restrict__ gate,
    float* x1_r, float* x1_i) {
  __shared__ float ar[4096];
  __shared__ float ai[4096];
  int bt = blockIdx.x >> 6;
  int d  = blockIdx.x & 63;
  long base = ((long)(bt * D_ + d)) * HW_;

  for (int idx = threadIdx.x; idx < 4096; idx += 256) {
    ar[idx] = xs_r[base + idx];
    ai[idx] = xs_i[base + idx];
  }
  __syncthreads();

  fft_dim(ar, ai, threadIdx.x, -1.f, 0);   // forward rows
  fft_dim(ar, ai, threadIdx.x, -1.f, 1);   // forward cols

  for (int idx = threadIdx.x; idx < 4096; idx += 256) {
    float wr = spec_wr[(long)d * HW_ + idx];
    float wi = spec_wi[(long)d * HW_ + idx];
    float r = ar[idx], im = ai[idx];
    ar[idx] = r * wr - im * wi;
    ai[idx] = r * wi + im * wr;
  }
  __syncthreads();

  fft_dim(ar, ai, threadIdx.x, 1.f, 0);    // inverse rows
  fft_dim(ar, ai, threadIdx.x, 1.f, 1);    // inverse cols

  float gg = gate[0];
  const float scale = 1.f / 4096.f;
  for (int idx = threadIdx.x; idx < 4096; idx += 256) {
    float sr = ar[idx] * scale, si = ai[idx] * scale;
    float cr = cliff_r_in[base + idx], ci = cliff_i_in[base + idx];
    x1_r[base + idx] = gg * cr + (1.f - gg) * sr + x0_r[base + idx];
    x1_i[base + idx] = gg * ci + (1.f - gg) * si + x0_i[base + idx];
  }
}

// ---------------------------------------------------------------------------
// Generic WMMA GEMM: C = act(A(f16, MxK row-major) x Bfrag + bias)
// NTILES: compile-time n-tile count per 128-wide block (8 or 4).
// mode 0: f32 row-major (encode)     mode 1: GELU -> f16 rows (MLP1)
// mode 2: drift + resid -> x2_r      mode 3: final output + resid -> d_out
// ---------------------------------------------------------------------------
template <int NTILES>
__global__ __launch_bounds__(256) void gemm_wmma(
    const half_t* __restrict__ A, const half_t* __restrict__ Bfrag,
    const float* __restrict__ bias, int M, int K, int N, int mode,
    float* Cf, half_t* Ch,
    const float* __restrict__ res_r, const float* __restrict__ res_i,
    float* __restrict__ outp) {
  int lane = threadIdx.x & 31, wid = threadIdx.x >> 5;
  int hi = lane >> 4;
  int m0 = blockIdx.x * 128 + wid * 16;
  int n0 = blockIdx.y * (NTILES * 16);
  int NT = N >> 4;
  int ktn = K >> 5;

  v8f acc[NTILES];
#pragma unroll
  for (int i = 0; i < NTILES; ++i) acc[i] = v8f{};

  const half_t* abase = A + (long)(m0 + (lane & 15)) * K;
  for (int kt = 0; kt < ktn; ++kt) {
    if (kt + 1 < ktn) __builtin_prefetch(abase + (kt + 1) * 32, 0, 0);
    v16h a = load_a_frag(abase + kt * 32, hi);
    const half_t* bptr = Bfrag + (((long)kt * NT + (n0 >> 4)) * 32 + lane) * 16;
#pragma unroll
    for (int nt = 0; nt < NTILES; ++nt) {
      v16h b = *(const v16h*)(bptr + (long)nt * 32 * 16);
      acc[nt] = wmma_f16(a, b, acc[nt]);
    }
  }

#pragma unroll
  for (int nt = 0; nt < NTILES; ++nt) {
    int n = n0 + nt * 16 + (lane & 15);
    float bv = bias ? bias[n] : 0.f;
#pragma unroll
    for (int v = 0; v < 8; ++v) {
      int m = m0 + v + 8 * hi;
      float val = acc[nt][v] + bv;
      if (mode == 0) {
        Cf[(long)m * N + n] = val;
      } else if (mode == 1) {
        float x3 = val * val * val;
        float gl = 0.5f * val *
                   (1.f + tanhf(0.7978845608028654f * (val + 0.044715f * x3)));
        Ch[(long)m * N + n] = (half_t)gl;
      } else if (mode == 2) {
        // m = (b*HW + hw)*T + t ; n = d   ->  x2_r[b,t,d,hw] = drift + x1_r
        int t = m & 15, hw = (m >> 4) & 4095, b = m >> 16;
        long idx = ((long)((b * T_ + t) * D_ + n)) * HW_ + hw;
        Cf[idx] = val + res_r[idx];
      } else {
        // m = bt*HW + hw ; n = feature (0..63 real, 64..127 imag)
        int hw = m & 4095, bt = m >> 12;
        int d = n & 63;
        long ridx = ((long)(bt * D_ + d)) * HW_ + hw;
        float rv = (n < 64) ? res_r[ridx] : res_i[ridx];
        outp[((long)(bt * D2_ + n)) * HW_ + hw] = val + rv;
      }
    }
  }
}

// ---------------------------------------------------------------------------
// Diagonal linear recurrence: wave per (b, hw), lane pair handles d, d+32.
// ---------------------------------------------------------------------------
__global__ __launch_bounds__(256) void recur_kernel(
    const float* __restrict__ xe, const float* __restrict__ lam_r,
    const float* __restrict__ lam_i, const float* __restrict__ dt,
    half_t* __restrict__ h_h) {
  int lane = threadIdx.x & 31, wid = threadIdx.x >> 5;
  int gwave = blockIdx.x * 8 + wid;        // b*HW + hw
  int b = gwave >> 12;
  long rowbase = (long)gwave * T_;
  float dtb = dt[b];

  float hr[2] = {0.f, 0.f}, hi2[2] = {0.f, 0.f};
  float dr[2], di[2], fr[2], fi[2];
#pragma unroll
  for (int s = 0; s < 2; ++s) {
    int d = lane + 32 * s;
    float l_r = lam_r[d], l_i = lam_i[d];
    float e = __expf(l_r * dtb);
    float sn, cs;
    __sincosf(l_i * dtb, &sn, &cs);
    dr[s] = e * cs; di[s] = e * sn;
    float nr = dr[s] - 1.f, ni = di[s];
    float den = l_r * l_r + l_i * l_i;
    fr[s] = (nr * l_r + ni * l_i) / den;
    fi[s] = (ni * l_r - nr * l_i) / den;
  }
  for (int t = 0; t < T_; ++t) {
    const float* row = xe + (rowbase + t) * D2_;
    half_t* orow = h_h + (rowbase + t) * D2_;
#pragma unroll
    for (int s = 0; s < 2; ++s) {
      int d = lane + 32 * s;
      float ur = row[d], ui = row[64 + d];
      float nhr = hr[s] * dr[s] - hi2[s] * di[s] + ur * fr[s] - ui * fi[s];
      float nhi = hr[s] * di[s] + hi2[s] * dr[s] + ur * fi[s] + ui * fr[s];
      hr[s] = nhr; hi2[s] = nhi;
      orow[d] = (half_t)nhr;
      orow[64 + d] = (half_t)nhi;
    }
  }
}

// ---------------------------------------------------------------------------
extern "C" void kernel_launch(void* const* d_in, const int* in_sizes, int n_in,
                              void* d_out, int out_size, void* d_ws, size_t ws_size,
                              hipStream_t stream) {
  (void)in_sizes; (void)n_in; (void)out_size; (void)ws_size;

  const float* x_real  = (const float*)d_in[0];
  const float* x_imag  = (const float*)d_in[1];
  const float* dt      = (const float*)d_in[2];
  const float* ln_s_g  = (const float*)d_in[3];
  const float* ln_s_b  = (const float*)d_in[4];
  const float* conv_w  = (const float*)d_in[5];
  const float* conv_b  = (const float*)d_in[6];
  const float* spec_wr = (const float*)d_in[7];
  const float* spec_wi = (const float*)d_in[8];
  const float* gate    = (const float*)d_in[9];
  const float* ln_t_g  = (const float*)d_in[10];
  const float* ln_t_b  = (const float*)d_in[11];
  const float* lam_r   = (const float*)d_in[12];
  const float* lam_i   = (const float*)d_in[13];
  const float* enc_r   = (const float*)d_in[14];
  const float* enc_i   = (const float*)d_in[15];
  const float* dec_r   = (const float*)d_in[16];
  const float* dec_i   = (const float*)d_in[17];
  const float* p_w1    = (const float*)d_in[18];
  const float* p_b1    = (const float*)d_in[19];
  const float* p_w2    = (const float*)d_in[20];
  const float* p_b2    = (const float*)d_in[21];
  float* out = (float*)d_out;

  char* ws = (char*)d_ws;
  size_t off = 0;
  auto take = [&](size_t bytes) -> char* {
    char* p = ws + off;
    off = (off + bytes + 255) & ~(size_t)255;
    return p;
  };
  float*  xs_r  = (float*)take(N5_ * 4);              // spatial LN out (real)
  float*  xs_i  = (float*)take(N5_ * 4);              // spatial LN out (imag)
  float*  x1_r  = (float*)take(N5_ * 4);              // cliff -> x1 (in place)
  float*  x1_i  = (float*)take(N5_ * 4);
  half_t* xt_h  = (half_t*)take((size_t)MROWS * D2_ * 2);  // temporal rows f16
  half_t* h_h   = (half_t*)take((size_t)MROWS * D2_ * 2);  // recurrence out f16
  float*  x2_r  = (float*)take(N5_ * 4);              // drift + resid (real)
  half_t* h1_h  = (half_t*)take((size_t)NPIX * F1_ * 2);   // MLP hidden f16
  float*  wenc  = (float*)take(D2_ * D2_ * 4);
  float*  wdec  = (float*)take(D2_ * D_ * 4);
  half_t* wc    = (half_t*)take((size_t)D2_ * KCONV * 2);
  half_t* bf_enc = (half_t*)take(D2_ * D2_ * 2);
  half_t* bf_dec = (half_t*)take(D2_ * D_ * 2);
  half_t* bf_w1  = (half_t*)take(D2_ * F1_ * 2);
  half_t* bf_w2  = (half_t*)take(F1_ * D2_ * 2);
  // Aliases: xe (262144x128 f32 = 128 MiB) reuses xs_r+xs_i (dead after FFT);
  //          xp (MLP input rows) reuses xt_h (dead after encode GEMM).
  float*  xe   = xs_r;
  half_t* xp_h = xt_h;

  // ---- weight packing (tiny) ----
  pack_convw<<<144, 256, 0, stream>>>(conv_w, wc);
  build_wmats<<<64, 256, 0, stream>>>(enc_r, enc_i, dec_r, dec_i, wenc, wdec);
  pack_bfrag<<<64,  256, 0, stream>>>(wenc, bf_enc, D2_, D2_);
  pack_bfrag<<<32,  256, 0, stream>>>(wdec, bf_dec, D2_, D_);
  pack_bfrag<<<256, 256, 0, stream>>>(p_w1, bf_w1, D2_, F1_);
  pack_bfrag<<<256, 256, 0, stream>>>(p_w2, bf_w2, F1_, D2_);

  dim3 gTile(BT_, HW_ / 64);

  // ---- stage 1: spatial ----
  cnorm_kernel<<<gTile, 256, 0, stream>>>(x_real, x_imag, ln_s_g, ln_s_b,
                                          1, 0, xs_r, xs_i, nullptr);
  conv3x3_wmma<<<BT_ * H_, 256, 0, stream>>>(xs_r, xs_i, wc, conv_b, x1_r, x1_i);
  fft_spec_combine<<<BT_ * D_, 256, 0, stream>>>(xs_r, xs_i, spec_wr, spec_wi,
                                                 x1_r, x1_i, x_real, x_imag, gate,
                                                 x1_r, x1_i);

  // ---- stage 2: temporal ----
  cnorm_kernel<<<gTile, 256, 0, stream>>>(x1_r, x1_i, ln_t_g, ln_t_b,
                                          1, 1, nullptr, nullptr, xt_h);
  gemm_wmma<8><<<dim3(MROWS / 128, 1), 256, 0, stream>>>(
      xt_h, bf_enc, nullptr, MROWS, D2_, D2_, 0,
      xe, nullptr, nullptr, nullptr, nullptr);
  recur_kernel<<<(B_ * HW_) / 8, 256, 0, stream>>>(xe, lam_r, lam_i, dt, h_h);
  gemm_wmma<4><<<dim3(MROWS / 128, 1), 256, 0, stream>>>(
      h_h, bf_dec, nullptr, MROWS, D2_, D_, 2,
      x2_r, nullptr, x1_r, nullptr, nullptr);

  // ---- stage 3: MLP ----
  cnorm_kernel<<<gTile, 256, 0, stream>>>(x2_r, x1_i, ln_t_g, ln_t_b,
                                          0, 2, nullptr, nullptr, xp_h);
  gemm_wmma<8><<<dim3(NPIX / 128, F1_ / 128), 256, 0, stream>>>(
      xp_h, bf_w1, p_b1, NPIX, D2_, F1_, 1,
      nullptr, h1_h, nullptr, nullptr, nullptr);
  gemm_wmma<8><<<dim3(NPIX / 128, 1), 256, 0, stream>>>(
      h1_h, bf_w2, p_b2, NPIX, F1_, D2_, 3,
      nullptr, nullptr, x2_r, x1_i, out);
}
